// SwinV2Block_51144470560964
// MI455X (gfx1250) — matrix operand order repaired and checked by reading, not compiled, not run on hardware
//
#include <hip/hip_runtime.h>
#include <hip/hip_bf16.h>
#include <cstdint>

typedef __bf16 bf16;
typedef __attribute__((ext_vector_type(16))) __bf16 v16bf;
typedef __attribute__((ext_vector_type(8)))  float  v8f;

typedef __attribute__((address_space(3))) const unsigned char lds_byte_t;

__device__ __forceinline__ v8f v8zero() {
    v8f r;
#pragma unroll
    for (int i = 0; i < 8; i++) r[i] = 0.0f;
    return r;
}

// byte offset of a __shared__ object from the wave's LDS base
__device__ __forceinline__ unsigned lds_off_of(const void* p) {
    return (unsigned)(uintptr_t)(const lds_byte_t*)p;
}

// CDNA5 async DMA: memory -> LDS, 16 bytes per lane, tracked by ASYNCcnt
__device__ __forceinline__ void async_ld_b128(unsigned lds_byte, unsigned goff_byte,
                                              const void* sbase) {
    asm volatile("global_load_async_to_lds_b128 %0, %1, %2 offset:0"
                 :: "v"(lds_byte), "v"(goff_byte),
                    "s"((unsigned long long)(uintptr_t)sbase)
                 : "memory");
}
__device__ __forceinline__ void wait_async_le4() {
    asm volatile("s_wait_asynccnt 4" ::: "memory");
}
__device__ __forceinline__ void wait_async_0() {
    asm volatile("s_wait_asynccnt 0" ::: "memory");
}

// ---------------------------------------------------------------------------
// f32 -> bf16 conversion (weights)
// ---------------------------------------------------------------------------
__global__ void f2bf_kernel(const float* __restrict__ in, bf16* __restrict__ out, long n) {
    long i = (long)blockIdx.x * blockDim.x + threadIdx.x;
    if (i < n) out[i] = (bf16)in[i];
}

// ---------------------------------------------------------------------------
// Precompute bias+mask table: bm[cls][head][M][N], cls = (wh==7)*2 + (ww==7).
// 4 * 12 * 64 * 64 = 196608 f32.
// ---------------------------------------------------------------------------
__device__ __forceinline__ int region_lbl(int c) { return c < 56 ? 0 : (c < 60 ? 1 : 2); }

__global__ void bm_kernel(const float* __restrict__ rpb, float* __restrict__ bm) {
    int e = blockIdx.x * 256 + threadIdx.x;          // < 196608
    int cls = e / 49152;
    int rem = e % 49152;
    int head = rem >> 12;
    int mn = rem & 4095;
    int M = mn >> 6, Nn = mn & 63;
    int qy = M >> 3, qx = M & 7, ky = Nn >> 3, kx = Nn & 7;
    int ridx = (qy - ky + 7) * 15 + (qx - kx + 7);
    int chH = (cls >> 1) & 1, chW = cls & 1;
    int lqh = chH ? region_lbl(56 + qy) : 0;
    int lkh = chH ? region_lbl(56 + ky) : 0;
    int lqw = chW ? region_lbl(56 + qx) : 0;
    int lkw = chW ? region_lbl(56 + kx) : 0;
    bool same = (lqh == lkh) && (lqw == lkw);
    bm[e] = rpb[ridx * 12 + head] + (same ? 0.0f : -1e9f);
}

// ---------------------------------------------------------------------------
// LayerNorm over C=384, optionally fused with roll(-4,-4) + window partition.
// One block (128 threads) per token row.
// ---------------------------------------------------------------------------
__global__ __launch_bounds__(128) void ln_win_kernel(const float* __restrict__ x,
                                                     const float* __restrict__ g,
                                                     const float* __restrict__ b,
                                                     bf16* __restrict__ out,
                                                     int windowed) {
    __shared__ float s1[128], s2[128];
    int t = threadIdx.x;
    long row = blockIdx.x;
    const float* xr = x + row * 384;
    float v0 = xr[t], v1 = xr[t + 128], v2 = xr[t + 256];
    s1[t] = v0 + v1 + v2;
    s2[t] = v0 * v0 + v1 * v1 + v2 * v2;
    __syncthreads();
    for (int s = 64; s > 0; s >>= 1) {
        if (t < s) { s1[t] += s1[t + s]; s2[t] += s2[t + s]; }
        __syncthreads();
    }
    float mean = s1[0] * (1.0f / 384.0f);
    float var  = s2[0] * (1.0f / 384.0f) - mean * mean;
    float rstd = rsqrtf(fmaxf(var, 0.0f) + 1e-5f);
    long dst = row;
    if (windowed) {
        int bi = (int)(row >> 12), l = (int)(row & 4095);
        int h = l >> 6, w = l & 63;
        int hp = (h - 4) & 63, wp = (w - 4) & 63;   // roll by -SHIFT
        int win = bi * 64 + (hp >> 3) * 8 + (wp >> 3);
        int n = (hp & 7) * 8 + (wp & 7);
        dst = (long)win * 64 + n;
    }
    bf16* o = out + dst * 384;
    o[t]       = (bf16)((v0 - mean) * rstd * g[t]       + b[t]);
    o[t + 128] = (bf16)((v1 - mean) * rstd * g[t + 128] + b[t + 128]);
    o[t + 256] = (bf16)((v2 - mean) * rstd * g[t + 256] + b[t + 256]);
}

// ---------------------------------------------------------------------------
// q/k L2-normalize per (token, head), fold 1/max(tau,1e-3) into q, emit bf16
// Q/K/V in [win][head][n][d] layout.  One wave per (row, head).
// ---------------------------------------------------------------------------
__global__ __launch_bounds__(256) void qknorm_kernel(const float* __restrict__ qkv,
                                                     const float* __restrict__ tau,
                                                     bf16* __restrict__ Qb,
                                                     bf16* __restrict__ Kb,
                                                     bf16* __restrict__ Vb) {
    int gw = blockIdx.x * 8 + (threadIdx.x >> 5);
    int d  = threadIdx.x & 31;
    long row = gw / 12;
    int head = gw % 12;
    const float* base = qkv + row * 1152 + head * 32 + d;
    float qv = base[0], kv = base[384], vv = base[768];
    float sq = qv * qv, sk = kv * kv;
#pragma unroll
    for (int m = 16; m >= 1; m >>= 1) {
        sq += __shfl_xor(sq, m, 32);
        sk += __shfl_xor(sk, m, 32);
    }
    float nq = fmaxf(sqrtf(sq), 1e-12f);
    float nk = fmaxf(sqrtf(sk), 1e-12f);
    float scale = 1.0f / fmaxf(tau[head], 1e-3f);
    long win = row >> 6;
    int n = (int)(row & 63);
    long o = ((win * 12 + head) * 64 + n) * 32 + d;
    Qb[o] = (bf16)(qv / nq * scale);
    Kb[o] = (bf16)(kv / nk);
    Vb[o] = (bf16)vv;
}

// ---------------------------------------------------------------------------
// Windowed attention: one block (4 waves) per (window, head).
// QK^T (WMMA) + precomputed bias/mask -> softmax -> PV (WMMA).
// Q/K/V staged to LDS via async DMA.
// ---------------------------------------------------------------------------
__global__ __launch_bounds__(128) void attn_kernel(const bf16* __restrict__ Qb,
                                                   const bf16* __restrict__ Kb,
                                                   const bf16* __restrict__ Vb,
                                                   const float* __restrict__ bm,
                                                   bf16* __restrict__ outA) {
    __shared__ __attribute__((aligned(32))) bf16 Qs[64 * 32];
    __shared__ __attribute__((aligned(32))) bf16 Ks[64 * 32];
    __shared__ __attribute__((aligned(32))) bf16 Vs[64 * 32];
    __shared__ float Ss[64 * 64];
    __shared__ bf16  Ps[64 * 64];

    int blk = blockIdx.x;
    int win = blk / 12, head = blk % 12;
    long base = (long)blk * (64 * 32);
    int tid = threadIdx.x;

    unsigned qL = lds_off_of(Qs), kL = lds_off_of(Ks), vL = lds_off_of(Vs);
#pragma unroll
    for (int it = 0; it < 2; it++) {
        int idx = tid + it * 128;            // 2048 bf16 per matrix = 256 b128
        unsigned go = (unsigned)(idx * 16);
        async_ld_b128(qL + idx * 16, go, Qb + base);
        async_ld_b128(kL + idx * 16, go, Kb + base);
        async_ld_b128(vL + idx * 16, go, Vb + base);
    }
    wait_async_0();
    __syncthreads();

    int wv = tid >> 5, lane = tid & 31, lr = lane & 15, lh = lane >> 4;

    int widx = win & 63, wh = widx >> 3, ww = widx & 7;
    int cls = ((wh == 7) ? 2 : 0) + ((ww == 7) ? 1 : 0);
    const float* BM = bm + (((long)cls * 12 + head) << 12);

    // A fragment (q rows wv*16 .. wv*16+15, K=0..31), contiguous k = 16*lh + s
    v16bf aq;
    {
        int row = wv * 16 + lr;
#pragma unroll
        for (int j = 0; j < 8; j++) {
            int k = lh * 16 + 2 * j;
            aq[2 * j]     = Qs[row * 32 + k];
            aq[2 * j + 1] = Qs[row * 32 + k + 1];
        }
    }

#pragma unroll
    for (int jt = 0; jt < 4; jt++) {
        int col = jt * 16 + lr;                 // key token index
        v16bf bk = *(const v16bf*)(Ks + col * 32 + lh * 16);
        v8f acc = v8zero();
        acc = __builtin_amdgcn_wmma_f32_16x16x32_bf16(false, aq, false, bk,
                                                      (short)0, acc, false, false);
#pragma unroll
        for (int r = 0; r < 8; r++) {
            int idx = (wv * 16 + r + 8 * lh) * 64 + col;   // (query M, key N)
            Ss[idx] = acc[r] + BM[idx];
        }
    }
    __syncthreads();

    // softmax: 2 threads per row, combine with one shfl
    {
        int row = tid >> 1, half = tid & 1;
        int b0 = row * 64 + half * 32;
        float mx = -3.4e38f;
        for (int c = 0; c < 32; c++) mx = fmaxf(mx, Ss[b0 + c]);
        mx = fmaxf(mx, __shfl_xor(mx, 1, 32));
        float sum = 0.0f;
        for (int c = 0; c < 32; c++) {
            float e = __expf(Ss[b0 + c] - mx);
            Ss[b0 + c] = e;
            sum += e;
        }
        sum += __shfl_xor(sum, 1, 32);
        float inv = 1.0f / sum;
        for (int c = 0; c < 32; c++) Ps[b0 + c] = (bf16)(Ss[b0 + c] * inv);
    }
    __syncthreads();

    v8f accO[2] = { v8zero(), v8zero() };
#pragma unroll
    for (int ks = 0; ks < 64; ks += 32) {
        v16bf ap;
        int row = wv * 16 + lr;
#pragma unroll
        for (int j = 0; j < 8; j++) {
            int k = ks + lh * 16 + 2 * j;
            ap[2 * j]     = Ps[row * 64 + k];
            ap[2 * j + 1] = Ps[row * 64 + k + 1];
        }
#pragma unroll
        for (int nt = 0; nt < 2; nt++) {
            int d = nt * 16 + lr;
            v16bf bv;
#pragma unroll
            for (int e = 0; e < 16; e++) bv[e] = Vs[(ks + lh * 16 + e) * 32 + d];
            accO[nt] = __builtin_amdgcn_wmma_f32_16x16x32_bf16(false, ap, false, bv,
                                                               (short)0, accO[nt], false, false);
        }
    }
#pragma unroll
    for (int nt = 0; nt < 2; nt++)
#pragma unroll
        for (int r = 0; r < 8; r++) {
            int M = wv * 16 + r + 8 * lh;
            int d = nt * 16 + lr;
            outA[((long)win * 64 + M) * 384 + head * 32 + d] = (bf16)accO[nt][r];
        }
}

// ---------------------------------------------------------------------------
// Tiled bf16 WMMA GEMM, 128x128 block tile, K staged 32-deep in LDS,
// double-buffered via GLOBAL_LOAD_ASYNC_TO_LDS_B128 (ASYNCcnt).
// MODE 0: qkv    -> f32 out + bias
// MODE 1: proj   -> un-window/un-shift scatter + shortcut add -> f32 x1
// MODE 2: fc1    -> bias + exact GELU -> bf16 out
// MODE 3: fc2    -> bias + residual(x1) -> f32 out (d_out)
// ---------------------------------------------------------------------------
template <int MODE>
__global__ __launch_bounds__(256) void gemm_kernel(const bf16* __restrict__ A,
                                                   const bf16* __restrict__ Bw,
                                                   const float* __restrict__ bias,
                                                   int M, int N, int K,
                                                   float* __restrict__ outF,
                                                   bf16* __restrict__ outB,
                                                   const float* __restrict__ extra) {
    __shared__ __attribute__((aligned(32))) bf16 As[2][128 * 32];
    __shared__ __attribute__((aligned(32))) bf16 Bs[2][32 * 128];

    int m0 = blockIdx.y * 128, n0 = blockIdx.x * 128;
    int tid = threadIdx.x;
    int wave = tid >> 5, lane = tid & 31, lr = lane & 15, lh = lane >> 4;
    int wm = wave >> 1, wn = wave & 1;   // 4x2 wave grid, each 32(M) x 64(N)

    unsigned asL = lds_off_of(&As[0][0]);
    unsigned bsL = lds_off_of(&Bs[0][0]);

    v8f acc[2][4];
#pragma unroll
    for (int i = 0; i < 2; i++)
#pragma unroll
        for (int j = 0; j < 4; j++) acc[i][j] = v8zero();

    // issue one K-slice (4 async b128 loads per thread: 2 for A, 2 for B)
    auto issue = [&](int p, int k0) {
#pragma unroll
        for (int it = 0; it < 2; it++) {
            int idx = tid + it * 256;
            int ar = idx >> 2, aseg = idx & 3;       // 4 uint4 per 32-elem A row
            unsigned aoff = (unsigned)((((long)(m0 + ar)) * K + k0 + aseg * 8) * 2);
            async_ld_b128(asL + (unsigned)((p * 4096 + idx * 8) * 2), aoff, A);
            int br = idx >> 4, bseg = idx & 15;      // 16 uint4 per 128-elem B row
            unsigned boff = (unsigned)((((long)(k0 + br)) * N + n0 + bseg * 8) * 2);
            async_ld_b128(bsL + (unsigned)((p * 4096 + idx * 8) * 2), boff, Bw);
        }
    };

    int nk = K >> 5;
    issue(0, 0);
    for (int ki = 0; ki < nk; ki++) {
        int p = ki & 1;
        if (ki + 1 < nk) {
            issue(p ^ 1, (ki + 1) * 32);
            wait_async_le4();     // first 4 (current buffer) complete, in order
        } else {
            wait_async_0();
        }
        __syncthreads();

        v16bf af[2], bfr[4];
#pragma unroll
        for (int mt = 0; mt < 2; mt++) {
            int row = wm * 32 + mt * 16 + lr;
#pragma unroll
            for (int j = 0; j < 8; j++) {
                int k = lh * 16 + 2 * j;
                af[mt][2 * j]     = As[p][row * 32 + k];
                af[mt][2 * j + 1] = As[p][row * 32 + k + 1];
            }
        }
#pragma unroll
        for (int nt = 0; nt < 4; nt++) {
            int col = wn * 64 + nt * 16 + lr;
#pragma unroll
            for (int e = 0; e < 16; e++) bfr[nt][e] = Bs[p][(lh * 16 + e) * 128 + col];
        }
#pragma unroll
        for (int mt = 0; mt < 2; mt++)
#pragma unroll
            for (int nt = 0; nt < 4; nt++)
                acc[mt][nt] = __builtin_amdgcn_wmma_f32_16x16x32_bf16(
                    false, af[mt], false, bfr[nt], (short)0, acc[mt][nt], false, false);
        __syncthreads();   // protect buffer p before it is refilled at ki+2
    }

#pragma unroll
    for (int mt = 0; mt < 2; mt++) {
#pragma unroll
        for (int nt = 0; nt < 4; nt++) {
#pragma unroll
            for (int r = 0; r < 8; r++) {
                long row = m0 + wm * 32 + mt * 16 + r + 8 * lh;
                int col = n0 + wn * 64 + nt * 16 + lr;
                float v = acc[mt][nt][r] + bias[col];
                if (MODE == 0) {
                    outF[row * N + col] = v;
                } else if (MODE == 1) {
                    int win = (int)(row >> 6);
                    int bi = win >> 6, wl = win & 63;
                    int n = (int)(row & 63);
                    int hp = (wl >> 3) * 8 + (n >> 3);
                    int wp = (wl & 7) * 8 + (n & 7);
                    int h = (hp + 4) & 63, w = (wp + 4) & 63;   // roll back +SHIFT
                    long dst = ((long)bi * 4096 + h * 64 + w) * 384 + col;
                    outF[dst] = v + extra[dst];
                } else if (MODE == 2) {
                    float ge = 0.5f * v * (1.0f + erff(v * 0.70710678118f));
                    outB[row * N + col] = (bf16)ge;
                } else {
                    long o = row * N + col;
                    outF[o] = v + extra[o];
                }
            }
        }
    }
}

// ---------------------------------------------------------------------------
// Workspace layout (bytes):
//   0          Wqkv  bf16   884736
//   884736     Wproj bf16   294912
//   1179648    Wfc1  bf16  1179648
//   2359296    Wfc2  bf16  1179648
//   3538944    Xw bf16 (50331648)      -- reused as AttnOut after qkv GEMM
//   53870592   qkvF f32 (301989888)    -- reused: x1 f32 (first 100663296)
//   154533888                           --        Hmid bf16 (201326592)
//   355860480  Qb bf16 (50331648)      -- reused as Xn2 after attention
//   406192128  Kb bf16 (50331648)
//   456523776  Vb bf16 (50331648)
//   506855424  bm f32 (786432)         -- bias+mask table
// total ~508 MB
// ---------------------------------------------------------------------------
extern "C" void kernel_launch(void* const* d_in, const int* in_sizes, int n_in,
                              void* d_out, int out_size, void* d_ws, size_t ws_size,
                              hipStream_t stream) {
    const float* x      = (const float*)d_in[0];
    const float* ln1_g  = (const float*)d_in[3];
    const float* ln1_b  = (const float*)d_in[4];
    const float* qkv_w  = (const float*)d_in[5];
    const float* qkv_b  = (const float*)d_in[6];
    const float* proj_w = (const float*)d_in[7];
    const float* proj_b = (const float*)d_in[8];
    const float* tau    = (const float*)d_in[9];
    const float* rpb    = (const float*)d_in[10];
    const float* ln2_g  = (const float*)d_in[11];
    const float* ln2_b  = (const float*)d_in[12];
    const float* fc1_w  = (const float*)d_in[13];
    const float* fc1_b  = (const float*)d_in[14];
    const float* fc2_w  = (const float*)d_in[15];
    const float* fc2_b  = (const float*)d_in[16];

    char* ws = (char*)d_ws;
    bf16*  Wqkv  = (bf16*)(ws + 0);
    bf16*  Wproj = (bf16*)(ws + 884736);
    bf16*  Wfc1  = (bf16*)(ws + 1179648);
    bf16*  Wfc2  = (bf16*)(ws + 2359296);
    bf16*  Xw    = (bf16*)(ws + 3538944);     // also AttnOut
    float* qkvF  = (float*)(ws + 53870592);
    float* x1    = (float*)(ws + 53870592);   // aliases qkvF region (after use)
    bf16*  Hmid  = (bf16*)(ws + 154533888);
    bf16*  Qb    = (bf16*)(ws + 355860480);
    bf16*  Kb    = (bf16*)(ws + 406192128);
    bf16*  Vb    = (bf16*)(ws + 456523776);
    float* bm    = (float*)(ws + 506855424);
    bf16*  Xn2   = Qb;                        // aliases Qb region (after attention)
    float* out   = (float*)d_out;

    // 1. weights -> bf16, bias+mask table
    f2bf_kernel<<<(442368 + 255) / 256, 256, 0, stream>>>(qkv_w, Wqkv, 442368);
    f2bf_kernel<<<(147456 + 255) / 256, 256, 0, stream>>>(proj_w, Wproj, 147456);
    f2bf_kernel<<<(589824 + 255) / 256, 256, 0, stream>>>(fc1_w, Wfc1, 589824);
    f2bf_kernel<<<(589824 + 255) / 256, 256, 0, stream>>>(fc2_w, Wfc2, 589824);
    bm_kernel<<<768, 256, 0, stream>>>(rpb, bm);

    // 2. LN1 + roll(-4,-4) + window partition -> bf16
    ln_win_kernel<<<65536, 128, 0, stream>>>(x, ln1_g, ln1_b, Xw, 1);

    // 3. qkv GEMM: [65536,384] x [384,1152]
    gemm_kernel<0><<<dim3(1152 / 128, 65536 / 128), 256, 0, stream>>>(
        Xw, Wqkv, qkv_b, 65536, 1152, 384, qkvF, nullptr, nullptr);

    // 4. l2norm q,k + tau fold -> bf16 Q/K/V
    qknorm_kernel<<<98304, 256, 0, stream>>>(qkvF, tau, Qb, Kb, Vb);

    // 5. windowed attention (writes AttnOut into Xw region)
    attn_kernel<<<12288, 128, 0, stream>>>(Qb, Kb, Vb, bm, Xw);

    // 6. proj GEMM + un-window/un-shift scatter + residual -> x1
    gemm_kernel<1><<<dim3(3, 512), 256, 0, stream>>>(
        Xw, Wproj, proj_b, 65536, 384, 384, x1, nullptr, x);

    // 7. LN2 -> bf16
    ln_win_kernel<<<65536, 128, 0, stream>>>(x1, ln2_g, ln2_b, Xn2, 0);

    // 8. fc1 GEMM + GELU -> bf16
    gemm_kernel<2><<<dim3(12, 512), 256, 0, stream>>>(
        Xn2, Wfc1, fc1_b, 65536, 1536, 384, nullptr, Hmid, nullptr);

    // 9. fc2 GEMM + residual -> f32 d_out
    gemm_kernel<3><<<dim3(3, 512), 256, 0, stream>>>(
        Hmid, Wfc2, fc2_b, 65536, 384, 1536, out, nullptr, x1);
}
